// GCN_Net_738734375374
// MI455X (gfx1250) — matrix operand order repaired and checked by reading, not compiled, not run on hardware
//
#include <hip/hip_runtime.h>
#include <hip/hip_bf16.h>

#define GN    100000   // nodes
#define GE    1250000  // edges
#define DIN   128
#define HID   64
#define DOUT  40
#define NT16  (GN / 16)  // 6250 row tiles (N divisible by 16)

typedef __attribute__((ext_vector_type(16))) _Float16 v16h;
typedef __attribute__((ext_vector_type(8)))  float    v8f;
typedef __attribute__((ext_vector_type(4)))  float    v4f;

// W1 packed: 16 frags (kt 0..3, nt 0..3) x 32 lanes x 16 f16 = 8192 f16 (16 KB)
// W2 packed:  6 frags (kt 0..1, nt 0..2) x 32 lanes x 16 f16 = 3072 f16 (6 KB)
#define PW1_ELEMS 8192
#define PW2_ELEMS 3072

// ---------------------------------------------------------------------------
// degree / normalization
// ---------------------------------------------------------------------------
__global__ __launch_bounds__(256) void gcn_deg_init(float* __restrict__ deg) {
    int i = blockIdx.x * blockDim.x + threadIdx.x;
    if (i < GN) deg[i] = 1.0f;  // self-loop
}

__global__ __launch_bounds__(256) void gcn_deg_edges(const int* __restrict__ dst,
                                                     float* __restrict__ deg) {
    int e = blockIdx.x * blockDim.x + threadIdx.x;
    if (e < GE) atomicAdd(&deg[dst[e]], 1.0f);
}

__global__ __launch_bounds__(256) void gcn_rsqrt(float* __restrict__ dinv) {
    int i = blockIdx.x * blockDim.x + threadIdx.x;
    if (i < GN) dinv[i] = rsqrtf(dinv[i]);
}

// ---------------------------------------------------------------------------
// prepack W1/W2 (f32 row-major) into WMMA B-fragment layout (f16):
//   pack[((frag*32 + lane) * 16) + j] = W[(kt*32 + (lane>=16?16:0) + j)][nt*16 + (lane&15)]
// ---------------------------------------------------------------------------
__global__ __launch_bounds__(256) void gcn_pack_w1(const float* __restrict__ W1,
                                                   _Float16* __restrict__ pW1) {
    int t = blockIdx.x * blockDim.x + threadIdx.x;   // 512 threads: frag*32+lane
    if (t >= 16 * 32) return;
    const int lane = t & 31;
    const int frag = t >> 5;                  // kt*4 + nt
    const int kt = frag >> 2, nt = frag & 3;
    const int kbl = kt * 32 + ((lane >> 4) << 4);
    const int n   = nt * 16 + (lane & 15);
    #pragma unroll
    for (int j = 0; j < 16; ++j)
        pW1[t * 16 + j] = (_Float16)W1[(size_t)(kbl + j) * HID + n];
}

__global__ __launch_bounds__(256) void gcn_pack_w2(const float* __restrict__ W2,
                                                   _Float16* __restrict__ pW2) {
    int t = blockIdx.x * blockDim.x + threadIdx.x;   // 192 threads
    if (t >= 6 * 32) return;
    const int lane = t & 31;
    const int frag = t >> 5;                  // kt*3 + nt
    const int kt = frag / 3, nt = frag % 3;
    const int kbl = kt * 32 + ((lane >> 4) << 4);
    const int n   = nt * 16 + (lane & 15);
    #pragma unroll
    for (int j = 0; j < 16; ++j)
        pW2[t * 16 + j] = (n < DOUT) ? (_Float16)W2[(size_t)(kbl + j) * DOUT + n]
                                     : (_Float16)0.0f;
}

// ---------------------------------------------------------------------------
// GEMM1: H1[N,64] = X[N,128] @ W1[128,64]  via v_wmma_f32_16x16x32_f16
// one wave -> 16 rows x 64 cols (4 accumulators), K in 4 steps of 32.
// B fragments staged in LDS (16 KB), read as 2x ds_load_b128 per fragment.
// ---------------------------------------------------------------------------
__global__ __launch_bounds__(256) void gcn_gemm1(const float* __restrict__ X,
                                                 const _Float16* __restrict__ pW1,
                                                 float* __restrict__ H1) {
    __shared__ _Float16 sW[PW1_ELEMS];        // 16 KB
    {   // cooperative stage: 1024 uint4, 4 per thread
        const uint4* gsrc = (const uint4*)pW1;
        uint4* ldst = (uint4*)sW;
        #pragma unroll
        for (int k = 0; k < 4; ++k)
            ldst[threadIdx.x + k * 256] = gsrc[threadIdx.x + k * 256];
    }
    __syncthreads();

    const int wave = blockIdx.x * (blockDim.x >> 5) + (threadIdx.x >> 5);
    const int lane   = threadIdx.x & 31;
    const int lhi    = lane >> 4;
    const int n_lane = lane & 15;

    if (wave < NT16) {                         // wave-uniform: EXEC all-ones inside
        const int m0   = wave * 16;
        const int mrow = m0 + n_lane;
        const float* arow = X + (size_t)mrow * DIN;

        v8f acc[4] = {};
        #pragma unroll
        for (int kt = 0; kt < 4; ++kt) {
            const int kb = kt * 32;
            if (kt < 3) __builtin_prefetch(arow + kb + 32, 0, 0);
            // A fragment: contiguous k = 16*lhi + s, four float4 loads
            const v4f x0 = *(const v4f*)(arow + kb + lhi * 16);
            const v4f x1 = *(const v4f*)(arow + kb + lhi * 16 + 4);
            const v4f x2 = *(const v4f*)(arow + kb + lhi * 16 + 8);
            const v4f x3 = *(const v4f*)(arow + kb + lhi * 16 + 12);
            v16h a;
            #pragma unroll
            for (int j = 0; j < 4; ++j) {
                a[j]      = (_Float16)x0[j];
                a[j + 4]  = (_Float16)x1[j];
                a[j + 8]  = (_Float16)x2[j];
                a[j + 12] = (_Float16)x3[j];
            }
            #pragma unroll
            for (int nt = 0; nt < 4; ++nt) {
                const v16h b = *(const v16h*)(sW + (((kt * 4 + nt) * 32 + lane) * 16));
                acc[nt] = __builtin_amdgcn_wmma_f32_16x16x32_f16(
                    false, a, false, b, (short)0, acc[nt], false, false);
            }
        }
        // C/D layout: lane<16 -> M=r, lane>=16 -> M=r+8 ; N = lane&15
        #pragma unroll
        for (int nt = 0; nt < 4; ++nt) {
            #pragma unroll
            for (int r = 0; r < 8; ++r) {
                const int row = m0 + r + lhi * 8;
                H1[(size_t)row * HID + nt * 16 + n_lane] = acc[nt][r];
            }
        }
    }
}

// ---------------------------------------------------------------------------
// agg1[i,c] = b1[c] + H1[i,c]*dinv[i]^2   (bias + self-loop), float4-wide
// ---------------------------------------------------------------------------
__global__ __launch_bounds__(256) void gcn_agg1_init(const float* __restrict__ H1,
                                                     const float* __restrict__ dinv,
                                                     const float* __restrict__ b1,
                                                     float* __restrict__ agg1) {
    int idx = blockIdx.x * blockDim.x + threadIdx.x;   // N*16 float4 chunks
    if (idx < GN * (HID / 4)) {
        const int i = idx >> 4, c4 = (idx & 15) << 2;
        const float d = dinv[i];
        const float w = d * d;
        const v4f h = *(const v4f*)(H1 + (size_t)i * HID + c4);
        const v4f b = *(const v4f*)(b1 + c4);
        v4f o;
        #pragma unroll
        for (int j = 0; j < 4; ++j) o[j] = b[j] + h[j] * w;
        *(v4f*)(agg1 + (size_t)i * HID + c4) = o;
    }
}

// edge scatter, layer 1: 16 threads/edge, float4 chunk each (coalesced)
__global__ __launch_bounds__(256) void gcn_scatter1(const int* __restrict__ src,
                                                    const int* __restrict__ dst,
                                                    const float* __restrict__ H1,
                                                    const float* __restrict__ dinv,
                                                    float* __restrict__ agg1) {
    int t = blockIdx.x * blockDim.x + threadIdx.x;
    if (t >= GE * 16) return;
    const int e  = t >> 4;
    const int c4 = (t & 15) << 2;
    const int s = src[e], d = dst[e];
    const float w = dinv[s] * dinv[d];
    const v4f h = *(const v4f*)(H1 + (size_t)s * HID + c4);
    float* op = agg1 + (size_t)d * HID + c4;
    atomicAdd(op + 0, h[0] * w);
    atomicAdd(op + 1, h[1] * w);
    atomicAdd(op + 2, h[2] * w);
    atomicAdd(op + 3, h[3] * w);
}

// ---------------------------------------------------------------------------
// GEMM2: T2[N,40] = relu(agg1)[N,64] @ W2[64,40], cols padded to 48 (3 tiles)
// ---------------------------------------------------------------------------
__global__ __launch_bounds__(256) void gcn_gemm2(const float* __restrict__ A,
                                                 const _Float16* __restrict__ pW2,
                                                 float* __restrict__ T2) {
    __shared__ _Float16 sW[PW2_ELEMS];        // 6 KB = 384 uint4
    {
        const uint4* gsrc = (const uint4*)pW2;
        uint4* ldst = (uint4*)sW;
        for (int i = threadIdx.x; i < PW2_ELEMS / 8; i += 256)
            ldst[i] = gsrc[i];
    }
    __syncthreads();

    const int wave = blockIdx.x * (blockDim.x >> 5) + (threadIdx.x >> 5);
    const int lane   = threadIdx.x & 31;
    const int lhi    = lane >> 4;
    const int n_lane = lane & 15;

    if (wave < NT16) {
        const int m0   = wave * 16;
        const int mrow = m0 + n_lane;
        const float* arow = A + (size_t)mrow * HID;

        v8f acc[3] = {};
        #pragma unroll
        for (int kt = 0; kt < 2; ++kt) {
            const int kb = kt * 32;
            const v4f x0 = *(const v4f*)(arow + kb + lhi * 16);
            const v4f x1 = *(const v4f*)(arow + kb + lhi * 16 + 4);
            const v4f x2 = *(const v4f*)(arow + kb + lhi * 16 + 8);
            const v4f x3 = *(const v4f*)(arow + kb + lhi * 16 + 12);
            v16h a;
            #pragma unroll
            for (int j = 0; j < 4; ++j) {     // relu fused into pack
                a[j]      = (_Float16)(x0[j] > 0.0f ? x0[j] : 0.0f);
                a[j + 4]  = (_Float16)(x1[j] > 0.0f ? x1[j] : 0.0f);
                a[j + 8]  = (_Float16)(x2[j] > 0.0f ? x2[j] : 0.0f);
                a[j + 12] = (_Float16)(x3[j] > 0.0f ? x3[j] : 0.0f);
            }
            #pragma unroll
            for (int nt = 0; nt < 3; ++nt) {
                const v16h b = *(const v16h*)(sW + (((kt * 3 + nt) * 32 + lane) * 16));
                acc[nt] = __builtin_amdgcn_wmma_f32_16x16x32_f16(
                    false, a, false, b, (short)0, acc[nt], false, false);
            }
        }
        #pragma unroll
        for (int nt = 0; nt < 3; ++nt) {
            const int n = nt * 16 + n_lane;
            if (n < DOUT) {
                #pragma unroll
                for (int r = 0; r < 8; ++r) {
                    const int row = m0 + r + lhi * 8;
                    T2[(size_t)row * DOUT + n] = acc[nt][r];
                }
            }
        }
    }
}

// out[i,c] = b2[c] + T2[i,c]*dinv[i]^2  (float4-wide: 10 chunks/row)
__global__ __launch_bounds__(256) void gcn_out_init(const float* __restrict__ T2,
                                                    const float* __restrict__ dinv,
                                                    const float* __restrict__ b2,
                                                    float* __restrict__ out) {
    int idx = blockIdx.x * blockDim.x + threadIdx.x;   // N*10 float4 chunks
    if (idx < GN * (DOUT / 4)) {
        const int i = idx / 10, c4 = (idx - i * 10) * 4;
        const float d = dinv[i];
        const float w = d * d;
        const v4f h = *(const v4f*)(T2 + (size_t)i * DOUT + c4);
        const v4f b = *(const v4f*)(b2 + c4);
        v4f o;
        #pragma unroll
        for (int j = 0; j < 4; ++j) o[j] = b[j] + h[j] * w;
        *(v4f*)(out + (size_t)i * DOUT + c4) = o;
    }
}

// edge scatter, layer 2: 10 threads/edge, float4 chunk each
__global__ __launch_bounds__(256) void gcn_scatter2(const int* __restrict__ src,
                                                    const int* __restrict__ dst,
                                                    const float* __restrict__ T2,
                                                    const float* __restrict__ dinv,
                                                    float* __restrict__ out) {
    int t = blockIdx.x * blockDim.x + threadIdx.x;
    if (t >= GE * 10) return;
    const int e  = t / 10;
    const int c4 = (t - e * 10) * 4;
    const int s = src[e], d = dst[e];
    const float w = dinv[s] * dinv[d];
    const v4f h = *(const v4f*)(T2 + (size_t)s * DOUT + c4);
    float* op = out + (size_t)d * DOUT + c4;
    atomicAdd(op + 0, h[0] * w);
    atomicAdd(op + 1, h[1] * w);
    atomicAdd(op + 2, h[2] * w);
    atomicAdd(op + 3, h[3] * w);
}

// ---------------------------------------------------------------------------
extern "C" void kernel_launch(void* const* d_in, const int* in_sizes, int n_in,
                              void* d_out, int out_size, void* d_ws, size_t ws_size,
                              hipStream_t stream) {
    const float* X   = (const float*)d_in[0];   // [N,128]
    const float* W1  = (const float*)d_in[1];   // [128,64]
    const float* b1  = (const float*)d_in[2];   // [64]
    const float* W2  = (const float*)d_in[3];   // [64,40]
    const float* b2  = (const float*)d_in[4];   // [40]
    const int*   ei  = (const int*)d_in[5];     // [2,E] flat
    const int*   src = ei;
    const int*   dst = ei + GE;
    float*       out = (float*)d_out;           // [N,40]

    // workspace (floats): dinv | H1 (aliased as T2) | agg1 | pW1 | pW2
    float* ws   = (float*)d_ws;
    float* dinv = ws;                               // GN floats
    float* H1   = ws + 100096;                      // GN*64, 256B-aligned offset
    float* agg1 = H1 + (size_t)GN * HID;            // GN*64
    _Float16* pW1 = (_Float16*)(agg1 + (size_t)GN * HID);  // 8192 f16
    _Float16* pW2 = pW1 + PW1_ELEMS;                        // 3072 f16
    float* T2   = H1;                               // reuse after scatter1 done

    const int B = 256;
    const int gemm_blocks = (NT16 + 7) / 8;         // 8 waves per block

    gcn_deg_init <<<(GN + B - 1) / B, B, 0, stream>>>(dinv);
    gcn_deg_edges<<<(GE + B - 1) / B, B, 0, stream>>>(dst, dinv);
    gcn_rsqrt    <<<(GN + B - 1) / B, B, 0, stream>>>(dinv);

    gcn_pack_w1  <<<2, B, 0, stream>>>(W1, pW1);
    gcn_pack_w2  <<<1, B, 0, stream>>>(W2, pW2);

    gcn_gemm1    <<<gemm_blocks, B, 0, stream>>>(X, pW1, H1);
    gcn_agg1_init<<<(GN * (HID / 4) + B - 1) / B, B, 0, stream>>>(H1, dinv, b1, agg1);
    gcn_scatter1 <<<(GE * 16 + B - 1) / B, B, 0, stream>>>(src, dst, H1, dinv, agg1);

    gcn_gemm2    <<<gemm_blocks, B, 0, stream>>>(agg1, pW2, T2);
    gcn_out_init <<<(GN * (DOUT / 4) + B - 1) / B, B, 0, stream>>>(T2, dinv, b2, out);
    gcn_scatter2 <<<(GE * 10 + B - 1) / B, B, 0, stream>>>(src, dst, T2, dinv, out);
}
